// TeacherModelV3_88072599372004
// MI455X (gfx1250) — matrix operand order repaired and checked by reading, not compiled, not run on hardware
//
#include <hip/hip_runtime.h>
#include <hip/hip_bf16.h>

// ---------------------------------------------------------------------------
// RPN head + top-k proposal decode for MI455X (gfx1250).
//
// Output depends only on batch 0:  sigmoid(w_obj @ feat[0]) -> top-1000 ->
// decode boxes.  Total traffic ~4 MiB (feat[0]) => memory bound (~0.2us at
// 23.3 TB/s).  The channel contraction runs on v_wmma_f32_16x16x32_bf16 so
// each feat element is read exactly once and multiplied on the matrix pipe.
// Top-k is a single-workgroup full bitonic sort of 16384 packed 64-bit keys
// in LDS (128 KB -- CDNA5 WGP has 320 KB), which exactly reproduces
// jax.lax.top_k ordering (desc score, asc index tie-break).
// ---------------------------------------------------------------------------

typedef __attribute__((ext_vector_type(16))) __bf16 v16bf;
typedef __attribute__((ext_vector_type(8)))  float  v8f;

#define C_DIM   256
#define H_DIM   64
#define W_DIM   64
#define S_DIM   (H_DIM * W_DIM)      // 4096 spatial positions
#define NA      3                    // anchors
#define NKEYS   (NA * S_DIM)         // 12288 candidate scores
#define NSORT   16384                // next pow2 for bitonic
#define TOPK_N  1000
#define NCHUNK  (C_DIM / 32)         // 8 WMMA K-chunks

// ---------------------------------------------------------------------------
// Kernel 1: objectness = sigmoid(w_obj @ feat[0] + b_obj) via WMMA, packed as
// 64-bit sort keys:  key = (float_bits(score) << 32) | ~flat_idx.
// Sigmoid outputs are in (0,1): positive floats => bit pattern is monotonic,
// and ~idx makes smaller indices win ties under a descending sort.
//
// Grid: 32 blocks x 256 threads (8 waves).  Each wave owns 16 spatial cols.
// ---------------------------------------------------------------------------
__global__ __launch_bounds__(256) void rpn_obj_wmma_kernel(
    const float* __restrict__ feat,     // [B, C, H, W]; we use batch 0
    const float* __restrict__ w_obj,    // [3, 256]
    const float* __restrict__ b_obj,    // [3]
    unsigned long long* __restrict__ keys) // [NKEYS]
{
    const int lane   = threadIdx.x & 31;
    const int wid    = threadIdx.x >> 5;
    const int n      = lane & 15;       // column within 16-wide tile
    const int hi     = lane >> 4;       // lane half (selects K sub-range)
    const int m      = lane & 15;       // A-matrix row (anchor; rows >=3 zeroed)
    const int s_base = blockIdx.x * 128 + wid * 16;
    const int s      = s_base + n;

    // --- A fragments: w_obj tile [16 x 32] bf16 per chunk --------------------
    // Loads are UNCONDITIONAL (clamped row) with zero-padding applied as a
    // value mask, so no EXEC-divergent load blocks; w_obj is K-loop invariant,
    // so preload all 8 fragments once (64 VGPRs).
    // ISA 16-bit A 16x32 layout: lane group 0 holds K = e+(e&8) (K 0-7,16-23),
    // lane group 1 holds +8 (K 8-15, 24-31).
    const float* wrow = w_obj + (size_t)((m < NA) ? m : 0) * C_DIM;
    const float  msk  = (m < NA) ? 1.0f : 0.0f;

    v16bf afrag[NCHUNK];
    #pragma unroll
    for (int c = 0; c < NCHUNK; ++c) {
        #pragma unroll
        for (int e = 0; e < 16; ++e) {
            int k = c * 32 + e + (e & 8) + 8 * hi;
            afrag[c][e] = (__bf16)(wrow[k] * msk);
        }
    }

    // --- K loop: 8 chunks of 32 channels -> 8 WMMAs (fully unrolled) --------
    // B layout: lanes 0-15 hold K=0-7,16-23, lanes 16-31 hold K=8-15,24-31; N=lane&15.
    // Each load is 64B-coalesced per half-wave (16 consecutive spatials).
    v8f acc = {};
    #pragma unroll
    for (int c = 0; c < NCHUNK; ++c) {
        v16bf bv;
        #pragma unroll
        for (int e = 0; e < 16; ++e) {
            int k = c * 32 + e + (e & 8) + 8 * hi;
            bv[e] = (__bf16)feat[(size_t)k * S_DIM + s];
        }
        acc = __builtin_amdgcn_wmma_f32_16x16x32_bf16(
            /*neg_a=*/false, afrag[c], /*neg_b=*/false, bv,
            /*c_mod=*/(short)0, acc, /*reuse_a=*/false, /*reuse_b=*/false);
    }

    // f32 16x16 C/D layout: lanes 0-15, VGPR r holds row M=r.  Anchors 0..2
    // therefore live in acc[0..2] of lanes 0-15.
    if (hi == 0) {
        #pragma unroll
        for (int a = 0; a < NA; ++a) {
            float logit = acc[a] + b_obj[a];
            float sc    = 1.0f / (1.0f + __expf(-logit));   // sigmoid, in (0,1)
            unsigned idxFlat = (unsigned)(a * S_DIM + s);   // reshape(B,-1) order
            unsigned sb = __float_as_uint(sc);
            keys[idxFlat] =
                ((unsigned long long)sb << 32) | (unsigned)(~idxFlat);
        }
    }
}

// ---------------------------------------------------------------------------
// Kernel 2: single-workgroup bitonic top-k + box decode.
// 1024 threads, 16384 u64 keys in 128 KB dynamic LDS (gfx1250 WGP: 320 KB).
// Descending sort; padding keys are 0 (< any real score bits) so they sink.
// ---------------------------------------------------------------------------
__global__ __launch_bounds__(1024) void topk_decode_kernel(
    const unsigned long long* __restrict__ keys,
    float* __restrict__ out)            // [TOPK_N, 5] = x1,y1,x2,y2,score
{
    extern __shared__ unsigned long long sk[];   // NSORT entries = 128 KB
    const int tid = threadIdx.x;

    for (int i = tid; i < NSORT; i += 1024)
        sk[i] = (i < NKEYS) ? keys[i] : 0ull;
    __syncthreads();

    // Bitonic sort, descending.
    for (int k = 2; k <= NSORT; k <<= 1) {
        for (int j = k >> 1; j > 0; j >>= 1) {
            for (int i = tid; i < NSORT; i += 1024) {
                int ixj = i ^ j;
                if (ixj > i) {
                    unsigned long long a = sk[i], b = sk[ixj];
                    bool desc = ((i & k) == 0);
                    if (desc ? (a < b) : (a > b)) { sk[i] = b; sk[ixj] = a; }
                }
            }
            __syncthreads();
        }
    }

    // Decode top-1000 exactly as the reference scalar math:
    //   anchor = idx % 3; spatial = idx / 3; y=(spatial/W)*16; x=(spatial%W)*16
    for (int i = tid; i < TOPK_N; i += 1024) {
        unsigned long long key = sk[i];
        float score  = __uint_as_float((unsigned)(key >> 32));
        unsigned idx = ~(unsigned)key;               // recover flat index
        int anchor  = (int)(idx % 3u);
        int spatial = (int)(idx / 3u);
        float y = (float)(spatial / W_DIM) * 16.0f;
        float x = (float)(spatial % W_DIM) * 16.0f;
        float sz = (anchor == 0) ? 64.0f : ((anchor == 1) ? 96.0f : 144.0f);
        float hf = sz * 0.5f;
        out[i * 5 + 0] = fmaxf(0.0f,    x - hf);
        out[i * 5 + 1] = fmaxf(0.0f,    y - hf);
        out[i * 5 + 2] = fminf(1024.0f, x + hf);
        out[i * 5 + 3] = fminf(1024.0f, y + hf);
        out[i * 5 + 4] = score;
    }
}

extern "C" void kernel_launch(void* const* d_in, const int* in_sizes, int n_in,
                              void* d_out, int out_size, void* d_ws, size_t ws_size,
                              hipStream_t stream) {
    (void)in_sizes; (void)n_in; (void)out_size; (void)ws_size;

    const float* feat  = (const float*)d_in[0];   // [32,256,64,64] f32
    const float* w_obj = (const float*)d_in[1];   // [3,256] f32
    const float* b_obj = (const float*)d_in[2];   // [3] f32
    // d_in[3]/d_in[4] (w_box/b_box) feed a dead value in the reference.

    unsigned long long* keys = (unsigned long long*)d_ws;  // 12288 * 8 B
    float* out = (float*)d_out;                            // [1000,5] f32

    // 4096 spatial / (8 waves * 16 cols) = 32 blocks.
    rpn_obj_wmma_kernel<<<32, 256, 0, stream>>>(feat, w_obj, b_obj, keys);

    // Single WGP: full bitonic top-k in 128 KB dynamic LDS.
    topk_decode_kernel<<<1, 1024, NSORT * sizeof(unsigned long long), stream>>>(
        keys, out);
}
